// QuantumAttention_47004122088021
// MI455X (gfx1250) — compile-verified, matrix operand order checked by reading, not run on hardware
//
#include <hip/hip_runtime.h>
#include <hip/hip_bf16.h>
#include <math.h>

// ---------------------------------------------------------------------------
// Types for CDNA5 WMMA (wave32): 16x16x32 bf16 -> f32
// ---------------------------------------------------------------------------
typedef __attribute__((ext_vector_type(16))) __bf16 v16bf;
typedef __attribute__((ext_vector_type(8)))  __bf16 v8bf;
typedef __attribute__((ext_vector_type(8)))  float  v8f;

#define HIDDEN 2048
#define SEQ    4096
#define HEADS  16
#define HDIM   128
#define KB     64   // attention key-block
#define LOG2E  1.4426950408889634f

#if __has_builtin(__builtin_amdgcn_tensor_load_to_lds)
#define HAS_TDM 1
#else
#define HAS_TDM 0
#endif

static __device__ inline float gelu_exact(float x) {
    return 0.5f * x * (1.0f + erff(x * 0.7071067811865475f));
}

// Generic->LDS address: low 32 bits of the generic address are the LDS offset.
static __device__ inline unsigned lds_addr(const void* p) {
    return (unsigned)(uintptr_t)p;
}

// CDNA5 async copy: LDS[lds_off..+15] = MEM[gptr..+15], tracked by ASYNCcnt.
static __device__ inline void async_ld_b128(unsigned lds_off, const void* gptr) {
    asm volatile("global_load_async_to_lds_b128 %0, %1, off"
                 :: "v"(lds_off), "v"((unsigned long long)(uintptr_t)gptr)
                 : "memory");
}
static __device__ inline void wait_async0() {
    asm volatile("s_wait_asynccnt 0x0" ::: "memory");
}

static __device__ inline void wait_tensor0() {
#if __has_builtin(__builtin_amdgcn_s_wait_tensorcnt)
    __builtin_amdgcn_s_wait_tensorcnt((short)0);
#else
    asm volatile("s_wait_tensorcnt 0x0" ::: "memory");
#endif
}

#if HAS_TDM
typedef __attribute__((ext_vector_type(4))) unsigned tdm_u32x4;
typedef __attribute__((ext_vector_type(8))) int      tdm_i32x8;
typedef __attribute__((ext_vector_type(4))) int      tdm_i32x4;

// One TDM 2D tile copy global->LDS; g1 carries dims/tile/stride/pad per ISA 8.4.
static __device__ inline void tdm_load_2d(unsigned lds_byte, const void* gsrc,
                                          tdm_i32x8 g1) {
    unsigned long long ga = (unsigned long long)(uintptr_t)gsrc;
    tdm_u32x4 g0 = { 1u,                                 // count=1 (valid D#)
                     lds_byte,                           // lds_addr
                     (unsigned)(ga & 0xffffffffu),       // global_addr[31:0]
                     ((unsigned)((ga >> 32) & 0x01ffffffu)) | 0x80000000u }; // type=2
    tdm_i32x4 z4 = { 0, 0, 0, 0 };
#if __clang_major__ >= 23
    tdm_i32x8 z8 = { 0, 0, 0, 0, 0, 0, 0, 0 };
    __builtin_amdgcn_tensor_load_to_lds(g0, g1, z4, z4, z8, 0);
#else
    __builtin_amdgcn_tensor_load_to_lds(g0, g1, z4, z4, 0);
#endif
}
#endif

// Load a 16x32 bf16 fragment (A or B operand) for v_wmma_f32_16x16x32_bf16.
// Lanes 0-15: rows 0-15 with K = 0..7 / 16..23; lanes 16-31: K = 8..15 / 24..31.
static __device__ inline v16bf frag_ld(const __bf16* base, int stride) {
    int lane = threadIdx.x & 31;
    const __bf16* p = base + (lane & 15) * stride + ((lane >> 4) << 3);
    v8bf lo = *(const v8bf*)p;
    v8bf hi = *(const v8bf*)(p + 16);
    return __builtin_shufflevector(lo, hi, 0, 1, 2, 3, 4, 5, 6, 7,
                                           8, 9, 10, 11, 12, 13, 14, 15);
}

// ---------------------------------------------------------------------------
// fp32 -> bf16 cast
// ---------------------------------------------------------------------------
__global__ void cast_f32_to_bf16(const float* __restrict__ in,
                                 __bf16* __restrict__ out, int n) {
    int i = blockIdx.x * blockDim.x + threadIdx.x;
    if (i < n) out[i] = (__bf16)in[i];
}

// ---------------------------------------------------------------------------
// GEMM (NT): C[M,N] = A[M,K] * B[N,K]^T, bf16 in, f32 accumulate.
// 256 threads (8 waves), 128x128 tile, K-step 32, async double-buffered LDS
// (ASYNCcnt path: global_load_async_to_lds_b128).
// MODE: 0 = f32 store, 1 = bf16 store, 2 = +bias+GELU bf16, 3 = +bias bf16,
//       4 = bf16 store transposed as V^T[h][d][s] (Cout[n*SEQ + m])
// ---------------------------------------------------------------------------
template <int MODE>
__global__ __launch_bounds__(256) void gemm_nt_bf16(
    const __bf16* __restrict__ A, const __bf16* __restrict__ B,
    int M, int N, int K, void* __restrict__ Cout,
    const float* __restrict__ bias) {
    __shared__ __bf16 As[2][128 * 40];   // 32 -> 40 stride: bank spread
    __shared__ __bf16 Bs[2][128 * 40];

    const int t    = threadIdx.x;
    const int lane = t & 31;
    const int wave = t >> 5;
    const int wr   = wave & 3;   // wave row (M)
    const int wc   = wave >> 2;  // wave col (N)
    const int mblk = blockIdx.y * 128;
    const int nblk = blockIdx.x * 128;

    v8f acc[2][4];
    for (int mi = 0; mi < 2; mi++)
        for (int ni = 0; ni < 4; ni++)
            for (int j = 0; j < 8; j++) acc[mi][ni][j] = 0.0f;

    const int lr = t >> 1;          // staged tile row (0..127)
    const int lc = (t & 1) << 4;    // staged tile col (0 or 16)
    const __bf16* ga = A + (size_t)(mblk + lr) * K + lc;
    const __bf16* gb = B + (size_t)(nblk + lr) * K + lc;

    auto stage = [&](int buf, int k0) {
        unsigned la = lds_addr(&As[buf][lr * 40 + lc]);
        unsigned lb = lds_addr(&Bs[buf][lr * 40 + lc]);
        async_ld_b128(la,      ga + k0);
        async_ld_b128(la + 16, ga + k0 + 8);
        async_ld_b128(lb,      gb + k0);
        async_ld_b128(lb + 16, gb + k0 + 8);
    };

    stage(0, 0);
    wait_async0();
    __syncthreads();

    int buf = 0;
    for (int k0 = 0; k0 < K; k0 += 32) {
        if (k0 + 32 < K) stage(buf ^ 1, k0 + 32);  // async engine runs under WMMA

        v16bf af[2], bfr[4];
        for (int mi = 0; mi < 2; mi++)
            af[mi] = frag_ld(&As[buf][(wr * 32 + mi * 16) * 40], 40);
        for (int ni = 0; ni < 4; ni++)
            bfr[ni] = frag_ld(&Bs[buf][(wc * 64 + ni * 16) * 40], 40);

        for (int mi = 0; mi < 2; mi++)
            for (int ni = 0; ni < 4; ni++)
                acc[mi][ni] = __builtin_amdgcn_wmma_f32_16x16x32_bf16(
                    false, af[mi], false, bfr[ni], (short)0, acc[mi][ni],
                    false, false);

        wait_async0();     // next tile landed in LDS
        __syncthreads();   // everyone done reading current tile
        buf ^= 1;
    }

    for (int mi = 0; mi < 2; mi++)
        for (int ni = 0; ni < 4; ni++)
            for (int j = 0; j < 8; j++) {
                int m = mblk + wr * 32 + mi * 16 + ((lane >> 4) << 3) + j;
                int n = nblk + wc * 64 + ni * 16 + (lane & 15);
                float v = acc[mi][ni][j];
                if (MODE == 0) {
                    ((float*)Cout)[(size_t)m * N + n] = v;
                } else if (MODE == 1) {
                    ((__bf16*)Cout)[(size_t)m * N + n] = (__bf16)v;
                } else if (MODE == 2) {
                    v = gelu_exact(v + bias[n]);
                    ((__bf16*)Cout)[(size_t)m * N + n] = (__bf16)v;
                } else if (MODE == 3) {
                    ((__bf16*)Cout)[(size_t)m * N + n] = (__bf16)(v + bias[n]);
                } else {  // MODE 4: V^T[h][d][s] = (h*128+d)*SEQ + s = n*SEQ + m
                    ((__bf16*)Cout)[(size_t)n * SEQ + m] = (__bf16)v;
                }
            }
}

// ---------------------------------------------------------------------------
// Fused flash attention, one (head, 128-query block) per workgroup.
// Each of 8 waves owns 16 complete query rows -> softmax row reductions are
// 16-lane shuffles. Q fragments pinned in VGPRs. K blocks AND pre-transposed
// V^T blocks are both copied by the Tensor Data Mover, double-buffered: wave 0
// issues two tensor_load_to_lds per key-block which complete under the
// previous block's S-GEMM + softmax + PV. One barrier per key-block.
// Vt layout: [HEADS][HDIM][SEQ] bf16 (produced by GEMM MODE 4).
// ---------------------------------------------------------------------------
__global__ __launch_bounds__(256) void attn_fused(
    const __bf16* __restrict__ Q, const __bf16* __restrict__ Km,
    const __bf16* __restrict__ Vt, __bf16* __restrict__ Out) {
    __shared__ __bf16 Ks[2][KB * 136];    // K rows: 64 x (128 + 8 pad)
    __shared__ __bf16 Vs[2][HDIM * 72];   // V^T:   128 dims x (64 keys + 8 pad)
    __shared__ __bf16 Ps[8 * 16 * 72];    // per-wave 16x64 P tiles

    const int t    = threadIdx.x;
    const int lane = t & 31;
    const int wave = t >> 5;
    const int h    = blockIdx.y;
    const int q0   = blockIdx.x * 128;
    const int halfsel = lane >> 4;        // C-layout: row = j + 8*halfsel

    const __bf16* Kh  = Km + (size_t)h * HDIM;             // row stride HIDDEN
    const __bf16* Vth = Vt + (size_t)h * HDIM * SEQ;       // row stride SEQ

    // stage K block + V^T block for key offset kb into buffer `buf`
    auto stage_kv = [&](int buf, int kb) {
#if HAS_TDM
        if (wave == 0) {
            // K: tile 128(dims) x 64(rows), row stride 2048; LDS rows 256B+16B pad
            tdm_i32x8 gk = { 0x07510000,  // 2B | pad_en | interval=5(64dw) | amt=3(4dw)
                             0x08000000,  // tensor_dim0 = 2048
                             0x10000000,  // tensor_dim1 = 4096
                             0x00800000,  // tile_dim0 = 128
                             64,          // tile_dim1 = 64
                             2048, 0, 0 };
            tdm_load_2d(lds_addr(&Ks[buf][0]), Kh + (size_t)kb * HIDDEN, gk);
            // V^T: tile 64(keys) x 128(dims), row stride 4096; rows 128B+16B pad
            tdm_i32x8 gv = { 0x07110000,  // 2B | pad_en | interval=4(32dw) | amt=3(4dw)
                             0x10000000,  // tensor_dim0 = 4096
                             0x00800000,  // tensor_dim1 = 128
                             0x00400000,  // tile_dim0 = 64
                             128,         // tile_dim1 = 128
                             4096, 0, 0 };
            tdm_load_2d(lds_addr(&Vs[buf][0]), Vth + kb, gv);
        }
#else
        {   // K: thread -> (row kr, 32-elem chunk kc)
            int kr = t >> 2, kc = (t & 3) << 5;
            const __bf16* src = Kh + (size_t)(kb + kr) * HIDDEN + kc;
            unsigned l = lds_addr(&Ks[buf][kr * 136 + kc]);
            async_ld_b128(l,      src);
            async_ld_b128(l + 16, src + 8);
            async_ld_b128(l + 32, src + 16);
            async_ld_b128(l + 48, src + 24);
        }
        {   // V^T: thread -> (dim vr, 32-key chunk vc)
            int vr = t >> 1, vc = (t & 1) << 5;
            const __bf16* src = Vth + (size_t)vr * SEQ + kb + vc;
            unsigned l = lds_addr(&Vs[buf][vr * 72 + vc]);
            async_ld_b128(l,      src);
            async_ld_b128(l + 16, src + 8);
            async_ld_b128(l + 32, src + 16);
            async_ld_b128(l + 48, src + 24);
        }
#endif
    };
    auto wait_kv = [&]() {
#if HAS_TDM
        wait_tensor0();
#else
        wait_async0();
#endif
    };

    // Q fragments: 16 rows x 128 dims = 4 K-steps, pinned in VGPRs.
    v16bf qf[4];
    const __bf16* qbase = Q + (size_t)(q0 + wave * 16) * HIDDEN + h * HDIM;
    for (int ks = 0; ks < 4; ks++) qf[ks] = frag_ld(qbase + ks * 32, HIDDEN);

    float rm[8], rl[8];
    v8f o[8];
    for (int j = 0; j < 8; j++) { rm[j] = -1e30f; rl[j] = 0.0f; }
    for (int nt = 0; nt < 8; nt++)
        for (int j = 0; j < 8; j++) o[nt][j] = 0.0f;

    const float scale2 = 0.08838834764831845f * LOG2E;  // (1/sqrt(128))*log2(e)
    __bf16* Pw = Ps + wave * 16 * 72;

    // prologue: block 0 resident before the loop
    stage_kv(0, 0);
    wait_kv();
    __syncthreads();

    int buf = 0;
    for (int kb = 0; kb < SEQ; kb += KB) {
        // TDM-prefetch the next K and V^T blocks; complete under this iteration
        if (kb + KB < SEQ) stage_kv(buf ^ 1, kb + KB);

        // ---- S = Q * K^T  (16 queries x 64 keys per wave) ----
        v8f sacc[4];
        for (int nt = 0; nt < 4; nt++)
            for (int j = 0; j < 8; j++) sacc[nt][j] = 0.0f;
        for (int ks = 0; ks < 4; ks++)
            for (int nt = 0; nt < 4; nt++) {
                v16bf kf = frag_ld(&Ks[buf][(nt * 16) * 136 + ks * 32], 136);
                sacc[nt] = __builtin_amdgcn_wmma_f32_16x16x32_bf16(
                    false, qf[ks], false, kf, (short)0, sacc[nt], false, false);
            }

        // ---- online softmax in base-2 (v_exp_f32 co-executes with WMMA) ----
        float mx[8];
        for (int j = 0; j < 8; j++) {
            float m = -1e30f;
            for (int nt = 0; nt < 4; nt++) m = fmaxf(m, sacc[nt][j]);
            for (int msk = 1; msk < 16; msk <<= 1)
                m = fmaxf(m, __shfl_xor(m, msk, 16));
            mx[j] = m * scale2;
        }
        for (int j = 0; j < 8; j++) {
            float mnew = fmaxf(rm[j], mx[j]);
            float corr = exp2f(rm[j] - mnew);
            float sum = 0.0f;
            int row = j + (halfsel << 3);
            for (int nt = 0; nt < 4; nt++) {
                float p = exp2f(sacc[nt][j] * scale2 - mnew);
                sum += p;
                Pw[row * 72 + nt * 16 + (lane & 15)] = (__bf16)p;
            }
            for (int msk = 1; msk < 16; msk <<= 1)
                sum += __shfl_xor(sum, msk, 16);
            rl[j] = rl[j] * corr + sum;
            rm[j] = mnew;
            for (int nt = 0; nt < 8; nt++) o[nt][j] *= corr;
        }

        // ---- O += P * V  (P is wave-private; same-wave LDS is in-order) ----
        for (int ks = 0; ks < 2; ks++) {
            v16bf pf = frag_ld(Pw + ks * 32, 72);
            for (int nt = 0; nt < 8; nt++) {
                v16bf vf = frag_ld(&Vs[buf][(nt * 16) * 72 + ks * 32], 72);
                o[nt] = __builtin_amdgcn_wmma_f32_16x16x32_bf16(
                    false, pf, false, vf, (short)0, o[nt], false, false);
            }
        }

        // ---- next K/V landed; single barrier releases both buffers ----
        wait_kv();
        __syncthreads();
        buf ^= 1;
    }

    // ---- normalize and write [seq, hidden] bf16 ----
    for (int j = 0; j < 8; j++) {
        float inv = 1.0f / rl[j];
        int row = q0 + wave * 16 + j + (halfsel << 3);
        for (int nt = 0; nt < 8; nt++) {
            int col = h * HDIM + nt * 16 + (lane & 15);
            Out[(size_t)row * HIDDEN + col] = (__bf16)(o[nt][j] * inv);
        }
    }
}

// ---------------------------------------------------------------------------
// Host orchestration
// ---------------------------------------------------------------------------
extern "C" void kernel_launch(void* const* d_in, const int* in_sizes, int n_in,
                              void* d_out, int out_size, void* d_ws,
                              size_t ws_size, hipStream_t stream) {
    (void)in_sizes; (void)n_in; (void)out_size; (void)ws_size;
    const float* hs = (const float*)d_in[0];
    const float* Wq = (const float*)d_in[1];
    const float* Wk = (const float*)d_in[2];
    const float* Wv = (const float*)d_in[3];
    const float* Wo = (const float*)d_in[4];
    const float* W1 = (const float*)d_in[5];
    const float* b1 = (const float*)d_in[6];
    const float* W2 = (const float*)d_in[7];
    const float* b2 = (const float*)d_in[8];

    char* ws = (char*)d_ws;
    size_t off = 0;
    auto alloc = [&](size_t elems) {
        __bf16* p = (__bf16*)(ws + off);
        off += ((elems * sizeof(__bf16)) + 255) & ~(size_t)255;
        return p;
    };
    const size_t SH = (size_t)SEQ * HIDDEN;     // 8M elems
    const size_t WW = (size_t)HIDDEN * HIDDEN;  // 4M elems
    __bf16* Xb  = alloc(SH);
    __bf16* Wqb = alloc(WW);
    __bf16* Wkb = alloc(WW);
    __bf16* Wvb = alloc(WW);
    __bf16* Wob = alloc(WW);
    __bf16* W1b = alloc(256 * 128);
    __bf16* W2b = alloc(128 * 256);
    __bf16* Qp  = alloc(SH);
    __bf16* Kp  = alloc(SH);
    __bf16* VtG = alloc(SH);                    // V^T [HEADS][HDIM][SEQ]
    __bf16* Hb  = alloc((size_t)SEQ * HEADS * 256);  // MLP hidden (reused)
    __bf16* Qm  = alloc(SH);
    __bf16* Km  = alloc(SH);
    __bf16* AO  = alloc(SH);

    auto cast = [&](const float* src, __bf16* dst, size_t n) {
        int blocks = (int)((n + 255) / 256);
        cast_f32_to_bf16<<<dim3(blocks), dim3(256), 0, stream>>>(src, dst, (int)n);
    };
    cast(hs, Xb, SH);
    cast(Wq, Wqb, WW); cast(Wk, Wkb, WW); cast(Wv, Wvb, WW); cast(Wo, Wob, WW);
    cast(W1, W1b, 256 * 128); cast(W2, W2b, 128 * 256);

    const dim3 blk(256);
    // projections (NT: both operands K-contiguous); V stored pre-transposed
    gemm_nt_bf16<1><<<dim3(HIDDEN / 128, SEQ / 128), blk, 0, stream>>>(
        Xb, Wqb, SEQ, HIDDEN, HIDDEN, Qp, nullptr);
    gemm_nt_bf16<1><<<dim3(HIDDEN / 128, SEQ / 128), blk, 0, stream>>>(
        Xb, Wkb, SEQ, HIDDEN, HIDDEN, Kp, nullptr);
    gemm_nt_bf16<4><<<dim3(HIDDEN / 128, SEQ / 128), blk, 0, stream>>>(
        Xb, Wvb, SEQ, HIDDEN, HIDDEN, VtG, nullptr);
    // per-head quantum MLP on Q: [65536,128] -> GELU[65536,256] -> [65536,128]
    gemm_nt_bf16<2><<<dim3(256 / 128, SEQ * HEADS / 128), blk, 0, stream>>>(
        Qp, W1b, SEQ * HEADS, 256, 128, Hb, b1);
    gemm_nt_bf16<3><<<dim3(128 / 128, SEQ * HEADS / 128), blk, 0, stream>>>(
        Hb, W2b, SEQ * HEADS, 128, 256, Qm, b2);
    // same for K (Hb reused)
    gemm_nt_bf16<2><<<dim3(256 / 128, SEQ * HEADS / 128), blk, 0, stream>>>(
        Kp, W1b, SEQ * HEADS, 256, 128, Hb, b1);
    gemm_nt_bf16<3><<<dim3(128 / 128, SEQ * HEADS / 128), blk, 0, stream>>>(
        Hb, W2b, SEQ * HEADS, 128, 256, Km, b2);
    // fused flash attention (TDM double-buffered K and V^T)
    attn_fused<<<dim3(SEQ / 128, HEADS), blk, 0, stream>>>(Qm, Km, VtG, AO);
    // output projection -> f32 result
    gemm_nt_bf16<0><<<dim3(HIDDEN / 128, SEQ / 128), blk, 0, stream>>>(
        AO, Wob, SEQ, HIDDEN, HIDDEN, d_out, nullptr);
}
